// FrequencyLinear_9113920602595
// MI455X (gfx1250) — matrix operand order repaired and checked by reading, not compiled, not run on hardware
//
#include <hip/hip_runtime.h>

// Complex linear out[b,o] = (x_re+i x_im)[b,:] . (w_re+i w_im)[o,:]
// Karatsuba: t1=xr*wr, t2=xi*wi, t3=(xr+xi)*(wr+wi); re=t1-t2, im=t3-t1-t2.
// Pass 1: one-time fp32->bf16 (+sum) pack into workspace.
// Pass 2: bf16 WMMA GEMM; double-buffered LDS fed by GLOBAL_LOAD_ASYNC_TO_LDS_B128
//         (ASYNCcnt-tracked, zero staging VGPRs), 1 barrier per K-step.

#define BATCH 4096
#define IN_F  2048
#define OUT_F 2048

#define BLK_M 128
#define BLK_N 64
#define BLK_K 32
#define LDSTR 20                   // dword row stride: 32 bf16 + pad, 16B-aligned, conflict-free

#define XSZ (BATCH * IN_F)         // 8388608 elements
#define WSZ (OUT_F * IN_F)         // 4194304 elements
#define WS_NEEDED ((size_t)(3 * XSZ + 3 * WSZ) * 2)   // 75,497,472 bytes

// per-buffer LDS dword offsets
#define O_ARE 0
#define O_AIM 2560
#define O_AS  5120
#define O_BRE 7680
#define O_BIM 8960
#define O_BS  10240
#define BUF_DW 11520               // 46080 B per buffer

#if __has_builtin(__builtin_amdgcn_global_load_async_to_lds_b128) && \
    __has_builtin(__builtin_amdgcn_s_wait_asynccnt)
#define USE_ASYNC 1
#else
#define USE_ASYNC 0
#endif

typedef __attribute__((ext_vector_type(16))) __bf16 v16bf;
typedef __attribute__((ext_vector_type(8)))  float  v8f;

#if USE_ASYNC
typedef int v4i __attribute__((ext_vector_type(4)));
static __device__ __forceinline__ __attribute__((address_space(1))) v4i* as_glb(const void* p) {
    return (__attribute__((address_space(1))) v4i*)(unsigned long long)p;
}
static __device__ __forceinline__ __attribute__((address_space(3))) v4i* as_lds(void* p) {
    // generic LDS pointer: low 32 bits are the LDS byte address
    return (__attribute__((address_space(3))) v4i*)(unsigned)(unsigned long long)p;
}
#endif

static __device__ __forceinline__ unsigned pack_bf2(float a, float b) {
    union { float f; unsigned u; } ua, ub;
    ua.f = a; ub.f = b;
    unsigned ra = (ua.u + 0x7FFFu + ((ua.u >> 16) & 1u)) >> 16;  // RNE
    unsigned rb = (ub.u + 0x7FFFu + ((ub.u >> 16) & 1u)) >> 16;
    return (ra & 0xFFFFu) | (rb << 16);
}

// ---------------- pass 1: fp32 -> bf16 {re, im, re+im} ----------------
__global__ __launch_bounds__(256)
void cvt_pack(const float* __restrict__ re, const float* __restrict__ im,
              unsigned short* __restrict__ ore, unsigned short* __restrict__ oim,
              unsigned short* __restrict__ osum, int n4) {
    const int i = blockIdx.x * blockDim.x + threadIdx.x;
    if (i >= n4) return;
    const float4 r = ((const float4*)re)[i];
    const float4 m = ((const float4*)im)[i];
    uint2 pr, pm, ps;
    pr.x = pack_bf2(r.x, r.y);             pr.y = pack_bf2(r.z, r.w);
    pm.x = pack_bf2(m.x, m.y);             pm.y = pack_bf2(m.z, m.w);
    ps.x = pack_bf2(r.x + m.x, r.y + m.y); ps.y = pack_bf2(r.z + m.z, r.w + m.w);
    ((uint2*)ore)[i] = pr;
    ((uint2*)oim)[i] = pm;
    ((uint2*)osum)[i] = ps;
}

// ---------------- WMMA fragment loads from LDS (explicit b128) ----------------
static __device__ __forceinline__ v16bf load_a_frag(const unsigned* S, int mBase, int lane) {
    const unsigned* p = S + (mBase + (lane & 15)) * LDSTR + ((lane >> 4) << 2);
    union { v16bf v; uint4 q[2]; } f;
    f.q[0] = *(const uint4*)(p);
    f.q[1] = *(const uint4*)(p + 8);
    return f.v;
}
static __device__ __forceinline__ v16bf load_b_frag(const unsigned* S, int nBase, int lane) {
    const unsigned* p = S + (nBase + (lane & 15)) * LDSTR + ((lane >> 4) << 2);
    union { v16bf v; uint4 q[2]; } f;
    f.q[0] = *(const uint4*)(p);
    f.q[1] = *(const uint4*)(p + 8);
    return f.v;
}

// ---------------- pass 2: bf16 complex GEMM ----------------
__global__ __launch_bounds__(256)
void cgemm_bf16(const unsigned short* __restrict__ xre, const unsigned short* __restrict__ xim,
                const unsigned short* __restrict__ xs,
                const unsigned short* __restrict__ wre, const unsigned short* __restrict__ wim,
                const unsigned short* __restrict__ ws,
                float* __restrict__ out) {
    extern __shared__ __attribute__((aligned(16))) unsigned smem[];  // 2*BUF_DW dwords

    const int tid  = threadIdx.x;
    const int lane = tid & 31;
    const int wid  = tid >> 5;
    const int wm   = wid >> 1;                  // 0..3
    const int wn   = wid & 1;                   // 0..1
    const int blkRow = blockIdx.y * BLK_M;
    const int blkCol = blockIdx.x * BLK_N;

    // staging: 4 x 16B chunks per 32-wide bf16 row; 64 rows per pass
    const int sc = tid & 3;                     // 16B chunk (8 bf16)
    const int sr = tid >> 2;                    // 0..63

    const unsigned short* gA[3] = { xre, xim, xs };
    const unsigned short* gB[3] = { wre, wim, ws };
    const int offA[3] = { O_ARE, O_AIM, O_AS };
    const int offB[3] = { O_BRE, O_BIM, O_BS };

    v8f t1[2][2] = {};  // xr*wr
    v8f t2[2][2] = {};  // xi*wi
    v8f t3[2][2] = {};  // (xr+xi)*(wr+wi)

    const int nIter = IN_F / BLK_K;             // 64

#if USE_ASYNC
    // memory -> LDS directly, no VGPR staging, tracked by ASYNCcnt
    auto async_tile = [&](int k0, unsigned* buf) {
#pragma unroll
        for (int a = 0; a < 3; ++a) {
#pragma unroll
            for (int p = 0; p < 2; ++p) {
                const int row = sr + (p << 6);
                const unsigned short* g = gA[a] + (size_t)(blkRow + row) * IN_F + k0 + (sc << 3);
                unsigned* l = buf + offA[a] + row * LDSTR + (sc << 2);
                __builtin_amdgcn_global_load_async_to_lds_b128(as_glb(g), as_lds(l), 0, 0);
            }
            const unsigned short* g = gB[a] + (size_t)(blkCol + sr) * IN_F + k0 + (sc << 3);
            unsigned* l = buf + offB[a] + sr * LDSTR + (sc << 2);
            __builtin_amdgcn_global_load_async_to_lds_b128(as_glb(g), as_lds(l), 0, 0);
        }
    };

    async_tile(0, smem);
    __builtin_amdgcn_s_wait_asynccnt(0);
    __syncthreads();
#else
    uint4 stA[3][2], stB[3];
    auto load_tile = [&](int k0) {
#pragma unroll
        for (int a = 0; a < 3; ++a) {
#pragma unroll
            for (int p = 0; p < 2; ++p) {
                const int row = sr + (p << 6);
                stA[a][p] = *(const uint4*)(gA[a] + (size_t)(blkRow + row) * IN_F + k0 + (sc << 3));
            }
            stB[a] = *(const uint4*)(gB[a] + (size_t)(blkCol + sr) * IN_F + k0 + (sc << 3));
        }
    };
    auto store_tile = [&](unsigned* buf) {
#pragma unroll
        for (int a = 0; a < 3; ++a) {
#pragma unroll
            for (int p = 0; p < 2; ++p) {
                const int row = sr + (p << 6);
                *(uint4*)(buf + offA[a] + row * LDSTR + (sc << 2)) = stA[a][p];
            }
            *(uint4*)(buf + offB[a] + sr * LDSTR + (sc << 2)) = stB[a];
        }
    };
#endif

    for (int i = 0; i < nIter; ++i) {
#if USE_ASYNC
        unsigned* buf = smem + (i & 1) * BUF_DW;
        if (i + 1 < nIter)
            async_tile((i + 1) * BLK_K, smem + ((i + 1) & 1) * BUF_DW);
#else
        unsigned* buf = smem;
        load_tile(i * BLK_K);
        store_tile(buf);
        __syncthreads();
#endif

        v16bf aRe[2], aIm[2], aS[2];
#pragma unroll
        for (int mt = 0; mt < 2; ++mt) {
            const int mb = wm * 32 + mt * 16;
            aRe[mt] = load_a_frag(buf + O_ARE, mb, lane);
            aIm[mt] = load_a_frag(buf + O_AIM, mb, lane);
            aS [mt] = load_a_frag(buf + O_AS,  mb, lane);
        }
#pragma unroll
        for (int nt = 0; nt < 2; ++nt) {
            const int nb = wn * 32 + nt * 16;
            const v16bf bRe = load_b_frag(buf + O_BRE, nb, lane);
            const v16bf bIm = load_b_frag(buf + O_BIM, nb, lane);
            const v16bf bS  = load_b_frag(buf + O_BS,  nb, lane);
#pragma unroll
            for (int mt = 0; mt < 2; ++mt) {
                t1[mt][nt] = __builtin_amdgcn_wmma_f32_16x16x32_bf16(
                    false, aRe[mt], false, bRe, (short)0, t1[mt][nt], false, false);
                t2[mt][nt] = __builtin_amdgcn_wmma_f32_16x16x32_bf16(
                    false, aIm[mt], false, bIm, (short)0, t2[mt][nt], false, false);
                t3[mt][nt] = __builtin_amdgcn_wmma_f32_16x16x32_bf16(
                    false, aS[mt],  false, bS,  (short)0, t3[mt][nt], false, false);
            }
        }

#if USE_ASYNC
        __builtin_amdgcn_s_wait_asynccnt(0);   // my async writes to next buffer done
        __syncthreads();                       // publish to all waves; retire reads of buf
#else
        __syncthreads();
#endif
    }

    // epilogue: re = t1 - t2 ; im = t3 - t1 - t2 ; interleaved (B, O, 2)
    const int mOff = ((lane >> 4) << 3);
    const int nLoc = lane & 15;
#pragma unroll
    for (int mt = 0; mt < 2; ++mt) {
#pragma unroll
        for (int nt = 0; nt < 2; ++nt) {
            const int col = blkCol + wn * 32 + nt * 16 + nLoc;
#pragma unroll
            for (int v = 0; v < 8; ++v) {
                const int row = blkRow + wm * 32 + mt * 16 + mOff + v;
                const float re = t1[mt][nt][v] - t2[mt][nt][v];
                const float im = t3[mt][nt][v] - t1[mt][nt][v] - t2[mt][nt][v];
                *(float2*)(out + (size_t)row * (OUT_F * 2) + (size_t)col * 2) =
                    make_float2(re, im);
            }
        }
    }
}

// ---------------- fallback: fused kernel (no workspace) ----------------
__global__ __launch_bounds__(256)
void cgemm_fused_f32(const float* __restrict__ x_re, const float* __restrict__ x_im,
                     const float* __restrict__ w_re, const float* __restrict__ w_im,
                     float* __restrict__ out) {
    __shared__ __attribute__((aligned(16))) unsigned sh[BUF_DW];
    const int tid = threadIdx.x, lane = tid & 31, wid = tid >> 5;
    const int wm = wid >> 1, wn = wid & 1;
    const int blkRow = blockIdx.y * BLK_M, blkCol = blockIdx.x * BLK_N;
    const int c4 = tid & 7, r0 = tid >> 3;

    v8f t1[2][2] = {}, t2[2][2] = {}, t3[2][2] = {};
    for (int k0 = 0; k0 < IN_F; k0 += BLK_K) {
#pragma unroll
        for (int p = 0; p < 4; ++p) {
            const int row = r0 + (p << 5);
            const size_t g = (size_t)(blkRow + row) * IN_F + k0 + (c4 << 2);
            const float4 xr = *(const float4*)(x_re + g);
            const float4 xi = *(const float4*)(x_im + g);
            const int li = row * LDSTR + (c4 << 1);
            sh[O_ARE + li] = pack_bf2(xr.x, xr.y); sh[O_ARE + li + 1] = pack_bf2(xr.z, xr.w);
            sh[O_AIM + li] = pack_bf2(xi.x, xi.y); sh[O_AIM + li + 1] = pack_bf2(xi.z, xi.w);
            sh[O_AS  + li] = pack_bf2(xr.x + xi.x, xr.y + xi.y);
            sh[O_AS  + li + 1] = pack_bf2(xr.z + xi.z, xr.w + xi.w);
        }
#pragma unroll
        for (int p = 0; p < 2; ++p) {
            const int row = r0 + (p << 5);
            const size_t g = (size_t)(blkCol + row) * IN_F + k0 + (c4 << 2);
            const float4 wr = *(const float4*)(w_re + g);
            const float4 wi = *(const float4*)(w_im + g);
            const int li = row * LDSTR + (c4 << 1);
            sh[O_BRE + li] = pack_bf2(wr.x, wr.y); sh[O_BRE + li + 1] = pack_bf2(wr.z, wr.w);
            sh[O_BIM + li] = pack_bf2(wi.x, wi.y); sh[O_BIM + li + 1] = pack_bf2(wi.z, wi.w);
            sh[O_BS  + li] = pack_bf2(wr.x + wi.x, wr.y + wi.y);
            sh[O_BS  + li + 1] = pack_bf2(wr.z + wi.z, wr.w + wi.w);
        }
        __syncthreads();
        v16bf aRe[2], aIm[2], aS[2];
#pragma unroll
        for (int mt = 0; mt < 2; ++mt) {
            const int mb = wm * 32 + mt * 16;
            aRe[mt] = load_a_frag(sh + O_ARE, mb, lane);
            aIm[mt] = load_a_frag(sh + O_AIM, mb, lane);
            aS [mt] = load_a_frag(sh + O_AS,  mb, lane);
        }
#pragma unroll
        for (int nt = 0; nt < 2; ++nt) {
            const int nb = wn * 32 + nt * 16;
            const v16bf bRe = load_b_frag(sh + O_BRE, nb, lane);
            const v16bf bIm = load_b_frag(sh + O_BIM, nb, lane);
            const v16bf bS  = load_b_frag(sh + O_BS,  nb, lane);
#pragma unroll
            for (int mt = 0; mt < 2; ++mt) {
                t1[mt][nt] = __builtin_amdgcn_wmma_f32_16x16x32_bf16(
                    false, aRe[mt], false, bRe, (short)0, t1[mt][nt], false, false);
                t2[mt][nt] = __builtin_amdgcn_wmma_f32_16x16x32_bf16(
                    false, aIm[mt], false, bIm, (short)0, t2[mt][nt], false, false);
                t3[mt][nt] = __builtin_amdgcn_wmma_f32_16x16x32_bf16(
                    false, aS[mt],  false, bS,  (short)0, t3[mt][nt], false, false);
            }
        }
        __syncthreads();
    }
    const int mOff = ((lane >> 4) << 3);
    const int nLoc = lane & 15;
#pragma unroll
    for (int mt = 0; mt < 2; ++mt)
#pragma unroll
        for (int nt = 0; nt < 2; ++nt) {
            const int col = blkCol + wn * 32 + nt * 16 + nLoc;
#pragma unroll
            for (int v = 0; v < 8; ++v) {
                const int row = blkRow + wm * 32 + mt * 16 + mOff + v;
                const float re = t1[mt][nt][v] - t2[mt][nt][v];
                const float im = t3[mt][nt][v] - t1[mt][nt][v] - t2[mt][nt][v];
                *(float2*)(out + (size_t)row * (OUT_F * 2) + (size_t)col * 2) =
                    make_float2(re, im);
            }
        }
}

extern "C" void kernel_launch(void* const* d_in, const int* in_sizes, int n_in,
                              void* d_out, int out_size, void* d_ws, size_t ws_size,
                              hipStream_t stream) {
    (void)in_sizes; (void)n_in; (void)out_size;
    const float* x_re = (const float*)d_in[0];
    const float* x_im = (const float*)d_in[1];
    const float* w_re = (const float*)d_in[2];
    const float* w_im = (const float*)d_in[3];
    float* out = (float*)d_out;

    dim3 grid(OUT_F / BLK_N, BATCH / BLK_M);   // (32, 32)
    dim3 block(256);

    if (ws_size >= WS_NEEDED) {
        unsigned short* p   = (unsigned short*)d_ws;
        unsigned short* xre = p;            p += XSZ;
        unsigned short* xim = p;            p += XSZ;
        unsigned short* xs  = p;            p += XSZ;
        unsigned short* wre = p;            p += WSZ;
        unsigned short* wim = p;            p += WSZ;
        unsigned short* wsm = p;

        cvt_pack<<<XSZ / 4 / 256, 256, 0, stream>>>(x_re, x_im, xre, xim, xs, XSZ / 4);
        cvt_pack<<<WSZ / 4 / 256, 256, 0, stream>>>(w_re, w_im, wre, wim, wsm, WSZ / 4);
        cgemm_bf16<<<grid, block, 2 * BUF_DW * sizeof(unsigned), stream>>>(
            xre, xim, xs, wre, wim, wsm, out);
    } else {
        cgemm_fused_f32<<<grid, block, 0, stream>>>(x_re, x_im, w_re, w_im, out);
    }
}
